// SE3Transformer_11690900979986
// MI455X (gfx1250) — compile-verified, matrix operand order checked by reading, not run on hardware
//
#include <hip/hip_runtime.h>

typedef __attribute__((ext_vector_type(16))) _Float16 v16h;
typedef __attribute__((ext_vector_type(8)))  float    v8f;

#define BSZ   2
#define NFINE 4096
#define NPC   1024
#define KDEG  8
#define BN    (BSZ*NFINE)      /* 8192  */
#define NE    (BN*KDEG)        /* 65536 */

// ---- workspace byte offsets ----
#define OFF_H0C  ((size_t)0)
#define OFF_H1C  (OFF_H0C  + (size_t)BN*32*4)   /* h0c [BN][32]     */
#define OFF_AGG0 (OFF_H1C  + (size_t)BN*96*4)   /* h1c [BN][32][3]  */
#define OFF_AGG1 (OFF_AGG0 + (size_t)BN*16*4)   /* agg0 [BN][16]    */
#define OFF_W2P  (OFF_AGG1 + (size_t)BN*48*4)   /* agg1 [BN][16][3] */
#define OFF_W3P  (OFF_W2P  + (size_t)4096*2)    /* packed f16 w2    */
/* OFF_W3P: packed f16 w3, 98304 halves */

__device__ __forceinline__ v8f wmma_f16(v16h a, v16h b) {
    v8f z = {};
    return __builtin_amdgcn_wmma_f32_16x16x32_f16(false, a, false, b,
                                                  (short)0, z, false, false);
}

// Scheduling fence: only VMEM-read (0x20) and DS-read (0x100) may cross, so
// the compiler can prefetch the next A tile but cannot interleave all WMMA
// chains (keeps live VGPRs < 256 -> no vgpr-MSB churn, better occupancy).
__device__ __forceinline__ void sched_fence() {
    __builtin_amdgcn_sched_barrier(0x120);
}

// ============================================================================
// Kernel 0: pack w2 (A-layout, 2 chunks of 16 outs) and w3 (A-layout, 16-col
// chunks) into f16 so the edge kernel loads WMMA A operands with 2x b128.
// Half index within a (col|m) row: t<8 -> K=8h+t ; t>=8 -> K=16+8h+(t-8).
// ============================================================================
__global__ __launch_bounds__(256) void prep_pack_kernel(
    const float* __restrict__ w2_00, const float* __restrict__ w2_01,
    const float* __restrict__ w2_10, const float* __restrict__ w2_11,
    const float* __restrict__ w3_00, const float* __restrict__ w3_01,
    const float* __restrict__ w3_10, const float* __restrict__ w3_11,
    _Float16* __restrict__ w2p, _Float16* __restrict__ w3p)
{
    int j = blockIdx.x * 256 + threadIdx.x;
    if (j < 4096) {                       // w2 pack: [p][chunk][m][h][t]
        int p  = j >> 10;
        int r  = j & 1023;
        int ch = r >> 9;
        int r2 = r & 511;
        int m  = r2 >> 5;
        int h  = (r2 >> 4) & 1;
        int t  = r2 & 15;
        int Kk = 8*h + ((t < 8) ? t : t + 8);
        const float* s = (p==0) ? w2_00 : (p==1) ? w2_01 : (p==2) ? w2_10 : w2_11;
        w2p[j] = (_Float16)s[Kk*32 + ch*16 + m];   // A[m=out][K=in] = w2[in][out]
    } else {                              // w3 pack: per path [col][h][t]
        int q = j - 4096;                 // 0..98303
        const float* s; int ncols; int qq;
        if      (q < 16384) { s = w3_00; ncols = 512;  qq = q;         }
        else if (q < 32768) { s = w3_01; ncols = 512;  qq = q - 16384; }
        else if (q < 49152) { s = w3_10; ncols = 512;  qq = q - 32768; }
        else                { s = w3_11; ncols = 1536; qq = q - 49152; }
        int col = qq >> 5;
        int h   = (qq >> 4) & 1;
        int t   = qq & 15;
        int Kk  = 8*h + ((t < 8) ? t : t + 8);
        w3p[q] = (_Float16)s[Kk*ncols + col];      // A[m=col][K=j] = w3[j][col]
    }
}

// ============================================================================
// Kernel 1: three-NN + inverse-squared-distance interpolation; builds
// h0c [BN][32] and h1c [BN][32][3] (interpolated 16ch ++ skip 16ch).
// ============================================================================
__global__ __launch_bounds__(256) void interp_kernel(
    const float* __restrict__ h0,   const float* __restrict__ h1,
    const float* __restrict__ uph0, const float* __restrict__ uph1,
    const float* __restrict__ xyz,  const float* __restrict__ xyzp,
    float* __restrict__ h0c, float* __restrict__ h1c)
{
    __shared__ float sxp[NPC * 3];
    int n = blockIdx.x * 256 + threadIdx.x;     // node 0..8191
    int b = n >> 12;
    const float* xpB = xyzp + (size_t)b * NPC * 3;
    for (int i = threadIdx.x; i < NPC * 3; i += 256) sxp[i] = xpB[i];
    __syncthreads();

    float px = xyz[n*3+0], py = xyz[n*3+1], pz = xyz[n*3+2];
    float bd0 = 3e38f, bd1 = 3e38f, bd2 = 3e38f;
    int   bi0 = 0,     bi1 = 0,     bi2 = 0;
    for (int j = 0; j < NPC; ++j) {
        float dx = px - sxp[j*3+0];
        float dy = py - sxp[j*3+1];
        float dz = pz - sxp[j*3+2];
        float dd = dx*dx + dy*dy + dz*dz;
        if (dd < bd0)      { bd2=bd1; bi2=bi1; bd1=bd0; bi1=bi0; bd0=dd; bi0=j; }
        else if (dd < bd1) { bd2=bd1; bi2=bi1; bd1=dd;  bi1=j; }
        else if (dd < bd2) { bd2=dd;  bi2=j; }
    }
    float w0 = 1.f/(fmaxf(bd0,1e-10f)+1e-8f);
    float w1 = 1.f/(fmaxf(bd1,1e-10f)+1e-8f);
    float w2 = 1.f/(fmaxf(bd2,1e-10f)+1e-8f);
    float inv = 1.f/(w0+w1+w2);
    w0 *= inv; w1 *= inv; w2 *= inv;

    const float* f0a = uph0 + (size_t)(b*NPC + bi0)*16;
    const float* f0b = uph0 + (size_t)(b*NPC + bi1)*16;
    const float* f0c = uph0 + (size_t)(b*NPC + bi2)*16;
    float* o0 = h0c + (size_t)n*32;
    #pragma unroll
    for (int c = 0; c < 16; ++c) o0[c]      = w0*f0a[c] + w1*f0b[c] + w2*f0c[c];
    #pragma unroll
    for (int c = 0; c < 16; ++c) o0[16 + c] = h0[(size_t)n*16 + c];

    const float* f1a = uph1 + (size_t)(b*NPC + bi0)*48;
    const float* f1b = uph1 + (size_t)(b*NPC + bi1)*48;
    const float* f1c = uph1 + (size_t)(b*NPC + bi2)*48;
    float* o1 = h1c + (size_t)n*96;
    #pragma unroll
    for (int c = 0; c < 48; ++c) o1[c]      = w0*f1a[c] + w1*f1b[c] + w2*f1c[c];
    #pragma unroll
    for (int c = 0; c < 48; ++c) o1[48 + c] = h1[(size_t)n*48 + c];
}

// ============================================================================
// Kernel 2: fused edge kernel. One wave = one 16-edge tile (= 2 dst nodes).
// radial-MLP (2 WMMAs/path) -> layer-3 WMMAs against packed W3, consuming each
// 16x16 R chunk from accumulator registers. Wave-local shfl_xor aggregation.
// launch_bounds(64, 4): cap at 256 arch VGPRs (4 waves/SIMD; LDS allows 8
// two-wave blocks per WGP at 36 KB each).
// ============================================================================
__global__ __launch_bounds__(64, 4) void edge_kernel(
    const int*   __restrict__ edge_src, const float* __restrict__ rbuf,
    const float* __restrict__ basis00,  const float* __restrict__ basis01,
    const float* __restrict__ basis10,  const float* __restrict__ basis11,
    const float* __restrict__ w1_00, const float* __restrict__ b1_00, const float* __restrict__ b2_00,
    const float* __restrict__ w1_01, const float* __restrict__ b1_01, const float* __restrict__ b2_01,
    const float* __restrict__ w1_10, const float* __restrict__ b1_10, const float* __restrict__ b2_10,
    const float* __restrict__ w1_11, const float* __restrict__ b1_11, const float* __restrict__ b2_11,
    const float* __restrict__ h0c, const float* __restrict__ h1c,
    const _Float16* __restrict__ w2p, const _Float16* __restrict__ w3p,
    float* __restrict__ agg0, float* __restrict__ agg1)
{
    __shared__ float tmpS[2][16 * 96 * 3];      // per-wave tmp11[e][i*3+f][m]

    const int wid  = threadIdx.x >> 5;
    const int lane = threadIdx.x & 31;
    const int e    = lane & 15;                 // edge within tile
    const int hh   = lane >> 4;                 // K/row half
    const int tile = blockIdx.x * 2 + wid;
    const int eg   = tile * 16 + e;             // global edge
    const int src  = edge_src[eg];
    const float rr = rbuf[eg];

    // ---- radial MLP via WMMA: build layer-3 B operands (f16, resident) ----
    const float* w1g[4] = {w1_00, w1_01, w1_10, w1_11};
    const float* b1g[4] = {b1_00, b1_01, b1_10, b1_11};
    const float* b2g[4] = {b2_00, b2_01, b2_10, b2_11};
    const v16h*  w2v    = (const v16h*)w2p;
    v16h Bx[4];
    #pragma unroll
    for (int p = 0; p < 4; ++p) {
        v16h B1;
        #pragma unroll
        for (int t = 0; t < 16; ++t) {
            int f = 8*hh + ((t < 8) ? t : t + 8);
            float v = fmaxf(rr * w1g[p][f] + b1g[p][f], 0.f);
            B1[t] = (_Float16)v;
        }
        v16h A0 = w2v[(p*2 + 0)*32 + e*2 + hh];
        v16h A1 = w2v[(p*2 + 1)*32 + e*2 + hh];
        v8f d0 = wmma_f16(A0, B1);              // outs 0..15
        v8f d1 = wmma_f16(A1, B1);              // outs 16..31
        v16h B2;
        #pragma unroll
        for (int t = 0; t < 8; ++t) {
            float v  = fmaxf(d0[t] + b2g[p][     8*hh + t], 0.f);
            float v2 = fmaxf(d1[t] + b2g[p][16 + 8*hh + t], 0.f);
            B2[t]     = (_Float16)v;            // K = 8*hh+t
            B2[t + 8] = (_Float16)v2;           // K = 16+8*hh+t
        }
        Bx[p] = B2;
        sched_fence();
    }

    // ---- register-resident per-edge operands for separable basis paths ----
    const float* h0row = h0c + (size_t)src * 32;
    const float* h1row = h1c + (size_t)src * 96;
    float b10r[3] = {basis10[eg*3+0], basis10[eg*3+1], basis10[eg*3+2]};
    float hs0r[16], tmp10r[16];
    #pragma unroll
    for (int idx = 0; idx < 16; ++idx) {
        int i = 8*hh + ((idx < 8) ? idx : idx + 8);
        hs0r[idx]   = h0row[i];
        tmp10r[idx] = b10r[0]*h1row[i*3+0] + b10r[1]*h1row[i*3+1] + b10r[2]*h1row[i*3+2];
    }
    float b00e    = basis00[eg];
    float b01r[3] = {basis01[eg*3+0], basis01[eg*3+1], basis01[eg*3+2]};
    sched_fence();

    // ---- tmp11[e][i*3+f][m] = sum_n basis11[m][n][f]*hs1[i][n] into LDS ----
    float bas[27];
    #pragma unroll
    for (int j = 0; j < 27; ++j) bas[j] = basis11[(size_t)eg*27 + j];
    float* T = &tmpS[wid][e * 288];
    #pragma unroll
    for (int ii = 0; ii < 16; ++ii) {
        int i = hh*16 + ii;
        float hn0 = h1row[i*3+0], hn1 = h1row[i*3+1], hn2 = h1row[i*3+2];
        #pragma unroll
        for (int f = 0; f < 3; ++f) {
            #pragma unroll
            for (int m = 0; m < 3; ++m) {
                T[(i*3 + f)*3 + m] =
                    bas[(m*3+0)*3+f]*hn0 + bas[(m*3+1)*3+f]*hn1 + bas[(m*3+2)*3+f]*hn2;
            }
        }
        if ((ii & 3) == 3) sched_fence();
    }
    __syncthreads();

    // ---- main loop: 12 WMMAs per output channel o ----
    const v16h* w3v   = (const v16h*)w3p;
    const v16h* A3_00 = w3v + 0;
    const v16h* A3_01 = w3v + 1024;
    const v16h* A3_10 = w3v + 2048;
    const v16h* A3_11 = w3v + 3072;
    const int n0 = tile * 2;

    for (int o = 0; o < 16; ++o) {
        if (o < 15)
            __builtin_prefetch(&A3_11[((o+1)*96 + e)*2 + hh], 0, 1);

        float m0 = 0.f, m1a = 0.f, m1b = 0.f, m1c = 0.f;
        #pragma unroll
        for (int sub = 0; sub < 2; ++sub) {     // paths 00/01/10, 32 cols each
            int aidx = ((o*32 + sub*16) + e)*2 + hh;
            v8f d00 = wmma_f16(A3_00[aidx], Bx[0]);
            v8f d01 = wmma_f16(A3_01[aidx], Bx[1]);
            v8f d10 = wmma_f16(A3_10[aidx], Bx[2]);
            float s00 = 0.f, s01 = 0.f, s10 = 0.f;
            #pragma unroll
            for (int v = 0; v < 8; ++v) {
                s00 += d00[v] * hs0r[sub*8 + v];
                s01 += d01[v] * hs0r[sub*8 + v];
                s10 += d10[v] * tmp10r[sub*8 + v];
            }
            m0  += b00e * s00 + s10;
            m1a += s01 * b01r[0];
            m1b += s01 * b01r[1];
            m1c += s01 * b01r[2];
            sched_fence();
        }
        #pragma unroll
        for (int sub = 0; sub < 6; ++sub) {     // path 11, 96 cols per o
            int aidx = ((o*96 + sub*16) + e)*2 + hh;
            v8f d = wmma_f16(A3_11[aidx], Bx[3]);
            int qb = sub*16 + 8*hh;
            #pragma unroll
            for (int v = 0; v < 8; ++v) {
                const float* tp = T + (qb + v)*3;
                float dv = d[v];
                m1a += dv * tp[0];
                m1b += dv * tp[1];
                m1c += dv * tp[2];
            }
            if (sub & 1) sched_fence();
        }
        // combine col halves (xor16) then sum 8 edges per node (xor4,2,1)
        m0  += __shfl_xor(m0, 16); m0  += __shfl_xor(m0, 4); m0  += __shfl_xor(m0, 2); m0  += __shfl_xor(m0, 1);
        m1a += __shfl_xor(m1a,16); m1a += __shfl_xor(m1a,4); m1a += __shfl_xor(m1a,2); m1a += __shfl_xor(m1a,1);
        m1b += __shfl_xor(m1b,16); m1b += __shfl_xor(m1b,4); m1b += __shfl_xor(m1b,2); m1b += __shfl_xor(m1b,1);
        m1c += __shfl_xor(m1c,16); m1c += __shfl_xor(m1c,4); m1c += __shfl_xor(m1c,2); m1c += __shfl_xor(m1c,1);
        if (hh == 0 && (lane & 7) == 0) {       // lanes 0 (node0) and 8 (node1)
            int node = n0 + (lane >> 3);
            agg0[node*16 + o] = m0;
            agg1[(node*16 + o)*3 + 0] = m1a;
            agg1[(node*16 + o)*3 + 1] = m1b;
            agg1[(node*16 + o)*3 + 2] = m1c;
        }
    }
}

// ============================================================================
// Kernel 3: out = agg/K + wself@h_c, then GNormSE3 per degree.
// ============================================================================
__global__ __launch_bounds__(256) void finalize_kernel(
    const float* __restrict__ agg0, const float* __restrict__ agg1,
    const float* __restrict__ h0c,  const float* __restrict__ h1c,
    const float* __restrict__ wself0, const float* __restrict__ wself1,
    const float* __restrict__ gamma0, const float* __restrict__ beta0,
    const float* __restrict__ gamma1, const float* __restrict__ beta1,
    float* __restrict__ out)
{
    __shared__ float sw0[512], sw1[512];
    for (int i = threadIdx.x; i < 512; i += 256) { sw0[i] = wself0[i]; sw1[i] = wself1[i]; }
    __syncthreads();

    int n = blockIdx.x * 256 + threadIdx.x;
    const float* h0row = h0c + (size_t)n * 32;
    const float* h1row = h1c + (size_t)n * 96;

    float o0[16], o1a[16], o1b[16], o1c[16];
    #pragma unroll
    for (int o = 0; o < 16; ++o) {
        float a  = agg0[n*16 + o] * 0.125f;
        float aa = agg1[(n*16 + o)*3 + 0] * 0.125f;
        float ab = agg1[(n*16 + o)*3 + 1] * 0.125f;
        float ac = agg1[(n*16 + o)*3 + 2] * 0.125f;
        #pragma unroll
        for (int i = 0; i < 32; ++i) {
            float ws0 = sw0[o*32 + i];
            float ws1 = sw1[o*32 + i];
            a  += ws0 * h0row[i];
            aa += ws1 * h1row[i*3 + 0];
            ab += ws1 * h1row[i*3 + 1];
            ac += ws1 * h1row[i*3 + 2];
        }
        o0[o] = a; o1a[o] = aa; o1b[o] = ab; o1c[o] = ac;
    }

    // gnorm degree-0
    float nn0[16], mu0 = 0.f;
    #pragma unroll
    for (int o = 0; o < 16; ++o) { nn0[o] = sqrtf(o0[o]*o0[o] + 1e-12f); mu0 += nn0[o]; }
    mu0 *= 0.0625f;
    float var0 = 0.f;
    #pragma unroll
    for (int o = 0; o < 16; ++o) { float d = nn0[o] - mu0; var0 += d*d; }
    var0 *= 0.0625f;
    float rs0 = rsqrtf(var0 + 1e-5f);
    #pragma unroll
    for (int o = 0; o < 16; ++o) {
        float t = fmaxf((nn0[o] - mu0) * rs0 * gamma0[o] + beta0[o], 0.f);
        out[n*16 + o] = t * (o0[o] / nn0[o]);
    }

    // gnorm degree-1
    float nn1[16], mu1 = 0.f;
    #pragma unroll
    for (int o = 0; o < 16; ++o) {
        nn1[o] = sqrtf(o1a[o]*o1a[o] + o1b[o]*o1b[o] + o1c[o]*o1c[o] + 1e-12f);
        mu1 += nn1[o];
    }
    mu1 *= 0.0625f;
    float var1 = 0.f;
    #pragma unroll
    for (int o = 0; o < 16; ++o) { float d = nn1[o] - mu1; var1 += d*d; }
    var1 *= 0.0625f;
    float rs1 = rsqrtf(var1 + 1e-5f);
    float* out1 = out + (size_t)BN * 16;
    #pragma unroll
    for (int o = 0; o < 16; ++o) {
        float t = fmaxf((nn1[o] - mu1) * rs1 * gamma1[o] + beta1[o], 0.f);
        float inv = t / nn1[o];
        out1[(n*16 + o)*3 + 0] = inv * o1a[o];
        out1[(n*16 + o)*3 + 1] = inv * o1b[o];
        out1[(n*16 + o)*3 + 2] = inv * o1c[o];
    }
}

// ============================================================================
extern "C" void kernel_launch(void* const* d_in, const int* in_sizes, int n_in,
                              void* d_out, int out_size, void* d_ws, size_t ws_size,
                              hipStream_t stream)
{
    (void)in_sizes; (void)n_in; (void)out_size; (void)ws_size;
    const float* h0    = (const float*)d_in[0];
    const float* h1    = (const float*)d_in[1];
    const float* uph0  = (const float*)d_in[2];
    const float* uph1  = (const float*)d_in[3];
    const float* xyz   = (const float*)d_in[4];
    const float* xyzp  = (const float*)d_in[5];
    const int*   esrc  = (const int*)d_in[6];
    const float* rr    = (const float*)d_in[8];
    const float* b00   = (const float*)d_in[9];
    const float* b01   = (const float*)d_in[10];
    const float* b10   = (const float*)d_in[11];
    const float* b11   = (const float*)d_in[12];
    const float* rw1[4], *rb1[4], *rw2[4], *rb2[4], *rw3[4];
    for (int p = 0; p < 4; ++p) {
        int base = 13 + p*5;
        rw1[p] = (const float*)d_in[base + 0];
        rb1[p] = (const float*)d_in[base + 1];
        rw2[p] = (const float*)d_in[base + 2];
        rb2[p] = (const float*)d_in[base + 3];
        rw3[p] = (const float*)d_in[base + 4];
    }
    const float* wself0 = (const float*)d_in[33];
    const float* wself1 = (const float*)d_in[34];
    const float* gamma0 = (const float*)d_in[35];
    const float* beta0  = (const float*)d_in[36];
    const float* gamma1 = (const float*)d_in[37];
    const float* beta1  = (const float*)d_in[38];
    (void)b11;

    char* ws = (char*)d_ws;
    float*    h0cW  = (float*)(ws + OFF_H0C);
    float*    h1cW  = (float*)(ws + OFF_H1C);
    float*    agg0W = (float*)(ws + OFF_AGG0);
    float*    agg1W = (float*)(ws + OFF_AGG1);
    _Float16* w2pW  = (_Float16*)(ws + OFF_W2P);
    _Float16* w3pW  = (_Float16*)(ws + OFF_W3P);

    prep_pack_kernel<<<400, 256, 0, stream>>>(rw2[0], rw2[1], rw2[2], rw2[3],
                                              rw3[0], rw3[1], rw3[2], rw3[3],
                                              w2pW, w3pW);
    interp_kernel<<<BN/256, 256, 0, stream>>>(h0, h1, uph0, uph1, xyz, xyzp, h0cW, h1cW);
    edge_kernel<<<(NE/16)/2, 64, 0, stream>>>(
        esrc, rr, b00, b01, b10, (const float*)d_in[12],
        rw1[0], rb1[0], rb2[0],
        rw1[1], rb1[1], rb2[1],
        rw1[2], rb1[2], rb2[2],
        rw1[3], rb1[3], rb2[3],
        h0cW, h1cW, w2pW, w3pW, agg0W, agg1W);
    finalize_kernel<<<BN/256, 256, 0, stream>>>(agg0W, agg1W, h0cW, h1cW,
                                                wself0, wself1,
                                                gamma0, beta0, gamma1, beta1,
                                                (float*)d_out);
}
